// GraphClassifier_74088185856644
// MI455X (gfx1250) — compile-verified, matrix operand order checked by reading, not run on hardware
//
#include <hip/hip_runtime.h>
#include <hip/hip_bf16.h>
#include <math.h>

// ---------------- problem constants (match reference) ----------------
#define NN     10000     // nodes == input feature width
#define H1D    200       // hidden dim
#define CD     8         // classes
#define NTEST  2000
#define KTILE  128       // K staged per LDS refill
#define NSTG   79        // ceil(NN / KTILE)
#define KP     (NSTG*KTILE)   // 10112: K padded for Bt (zero fill)
#define NP     208       // H1 padded to multiple of 16 (13 tiles)
#define LDSROW 136       // LDS row stride in bf16 elems (272 B, 16B aligned, conflict-skewed)

typedef __attribute__((ext_vector_type(8)))  float          v8f;
typedef __attribute__((ext_vector_type(4)))  float          v4f;
typedef __attribute__((ext_vector_type(16))) __bf16         v16bf;
typedef __attribute__((ext_vector_type(8)))  __bf16         v8bf;
typedef __attribute__((ext_vector_type(8)))  unsigned short v8us;

union ABOp { v16bf v; struct { v8bf lo; v8bf hi; } p; };

__device__ __forceinline__ unsigned short f32_to_bf16_rne(float f) {
    unsigned int u = __float_as_uint(f);
    u += 0x7FFFu + ((u >> 16) & 1u);
    return (unsigned short)(u >> 16);
}

// ---------------- small utility kernels ----------------
__global__ void zero_f32(float* __restrict__ p, int n) {
    int i = blockIdx.x * 256 + threadIdx.x;
    if (i < n) p[i] = 0.0f;
}

__global__ void deg_kernel(const int* __restrict__ ei, const float* __restrict__ ew,
                           float* __restrict__ deg, int E) {
    int e = blockIdx.x * 256 + threadIdx.x;
    if (e >= E + NN) return;
    if (e < E) atomicAdd(&deg[ei[E + e]], ew[e]);   // ei[1][e] = target
    else       atomicAdd(&deg[e - E], 1.0f);
}

__global__ void dis_kernel(float* __restrict__ deg) {
    int i = blockIdx.x * 256 + threadIdx.x;
    if (i < NN) { float d = deg[i]; deg[i] = (d > 0.0f) ? rsqrtf(d) : 0.0f; }
}

__global__ void norm_kernel(const int* __restrict__ ei, const float* __restrict__ ew,
                            const float* __restrict__ dis, float* __restrict__ nrm, int E) {
    int e = blockIdx.x * 256 + threadIdx.x;
    if (e >= E + NN) return;
    if (e < E) nrm[e] = dis[ei[e]] * ew[e] * dis[ei[E + e]];
    else { int i = e - E; nrm[e] = dis[i] * dis[i]; }
}

// W1 (NN x H1 row-major f32) -> Bt (NP x KP column-major bf16, zero padded)
__global__ void fill_bt(const float* __restrict__ W1, unsigned short* __restrict__ Bt) {
    int idx = blockIdx.x * 256 + threadIdx.x;
    if (idx >= NP * KP) return;
    int n = idx / KP;
    int k = idx - n * KP;
    float v = (n < H1D && k < NN) ? W1[(size_t)k * H1D + n] : 0.0f;
    Bt[idx] = f32_to_bf16_rne(v);
}

// ---------------- main WMMA GEMM: h1 = x @ W1  (bf16 in, f32 acc) ----------------
// grid.x = NN/16 (625) row blocks; block = 13 waves, wave w owns N-tile w.
// Waves 0-7 double-buffer stage x -> LDS bf16; all 13 waves run WMMA.
__global__ void __launch_bounds__(416)
gemm1_wmma(const float* __restrict__ x, const unsigned short* __restrict__ Bt,
           float* __restrict__ h1) {
    __shared__ __align__(16) unsigned short ldsA[2][16 * LDSROW];

    const int tid   = threadIdx.x;
    const int wave  = tid >> 5;               // 0..12 -> N tile
    const int lane  = tid & 31;
    const int mBase = blockIdx.x * 16;
    const int m     = lane & 15;              // A row within tile / C column lane
    const int koff  = (lane >> 4) << 3;       // 0 or 8 (16-bit operand K grouping)
    const int ncol  = wave * 16 + m;          // B column for this lane (< NP, padded)

    // staging role: 256 threads, one 8-float chunk each (16 rows x 16 chunks)
    const bool stager = (tid < 256);          // wave-uniform (waves 0..7)
    const int  smm    = tid >> 4;             // row 0..15
    const int  skk    = (tid & 15) << 3;      // k offset 0..120
    const float* xrow = x + (size_t)(mBase + smm) * NN;

    v8f acc = {};
    const __bf16* btRow = (const __bf16*)Bt + (size_t)ncol * KP;

    v4f r0 = {}, r1 = {};
    // prologue: fetch tile 0
    if (stager && skk < NN) {                 // always true for tile 0
        r0 = __builtin_nontemporal_load((const v4f*)(xrow + skk));
        r1 = __builtin_nontemporal_load((const v4f*)(xrow + skk + 4));
    }
    if (stager) {
        v8us pk;
        #pragma unroll
        for (int i = 0; i < 4; ++i) { pk[i] = f32_to_bf16_rne(r0[i]); pk[4 + i] = f32_to_bf16_rne(r1[i]); }
        *(v8us*)&ldsA[0][smm * LDSROW + skk] = pk;
    }
    __syncthreads();

    for (int t = 0; t < NSTG; ++t) {
        const int k0 = t * KTILE;
        // issue global loads for tile t+1 (overlapped with WMMA below)
        if (t + 1 < NSTG) {
            const int gk = (t + 1) * KTILE + skk;
            r0 = (v4f)0.0f; r1 = (v4f)0.0f;
            if (stager && gk < NN) {          // chunks are 8-aligned; NN%8==0 -> all-or-nothing
                r0 = __builtin_nontemporal_load((const v4f*)(xrow + gk));
                r1 = __builtin_nontemporal_load((const v4f*)(xrow + gk + 4));
            }
        }
        // 4 x WMMA 16x16x32 from current buffer
        const unsigned short* cur = ldsA[t & 1];
        #pragma unroll
        for (int s = 0; s < KTILE / 32; ++s) {
            const int kb = s * 32;
            ABOp a, b;
            a.p.lo = *(const v8bf*)&cur[m * LDSROW + kb + koff];
            a.p.hi = *(const v8bf*)&cur[m * LDSROW + kb + 16 + koff];
            b.p.lo = *(const v8bf*)(btRow + (k0 + kb + koff));
            b.p.hi = *(const v8bf*)(btRow + (k0 + kb + 16 + koff));
            acc = __builtin_amdgcn_wmma_f32_16x16x32_bf16(
                      false, a.v, false, b.v, (short)0, acc, false, false);
        }
        // convert + store tile t+1 into the other buffer
        if (t + 1 < NSTG && stager) {
            v8us pk;
            #pragma unroll
            for (int i = 0; i < 4; ++i) { pk[i] = f32_to_bf16_rne(r0[i]); pk[4 + i] = f32_to_bf16_rne(r1[i]); }
            *(v8us*)&ldsA[(t + 1) & 1][smm * LDSROW + skk] = pk;
        }
        __syncthreads();
    }

    // C/D layout: VGPR v -> M = v + 8*(lane>=16), N = lane&15 (within tile)
    const int mrow0 = (lane >> 4) << 3;
    if (ncol < H1D) {
        #pragma unroll
        for (int v = 0; v < 8; ++v)
            h1[(size_t)(mBase + mrow0 + v) * H1D + ncol] = acc[v];
    }
}

// ---------------- scatter-aggregate layer 1 (200 feats/edge) ----------------
__global__ void scatter200(const int* __restrict__ ei, const float* __restrict__ nrm,
                           const float* __restrict__ h, float* __restrict__ agg, int E) {
    int e = blockIdx.x;
    int f = threadIdx.x;
    if (f >= H1D) return;
    int r, c;
    if (e < E) { r = ei[e]; c = ei[E + e]; } else { r = c = e - E; }
    float w = nrm[e];
    atomicAdd(&agg[(size_t)c * H1D + f], h[(size_t)r * H1D + f] * w);
}

// ---------------- layer 2 linear, fused relu(agg1 + b1) ----------------
__global__ void gemm2_fused(const float* __restrict__ agg1, const float* __restrict__ b1,
                            const float* __restrict__ W2, float* __restrict__ h2) {
    int idx = blockIdx.x * 256 + threadIdx.x;
    if (idx >= NN * CD) return;
    int n = idx >> 3, c = idx & 7;
    const float* zr = agg1 + (size_t)n * H1D;
    float s = 0.0f;
    for (int k = 0; k < H1D; ++k) {
        float z = zr[k] + b1[k];
        z = (z > 0.0f) ? z : 0.0f;
        s += z * W2[k * CD + c];
    }
    h2[idx] = s;
}

// ---------------- scatter-aggregate layer 2 (8 feats/edge) ----------------
__global__ void scatter8(const int* __restrict__ ei, const float* __restrict__ nrm,
                         const float* __restrict__ h2, float* __restrict__ agg2, int E) {
    int idx = blockIdx.x * 256 + threadIdx.x;
    if (idx >= (E + NN) * CD) return;
    int e = idx >> 3, c8 = idx & 7;
    int r, c;
    if (e < E) { r = ei[e]; c = ei[E + e]; } else { r = c = e - E; }
    atomicAdd(&agg2[(size_t)c * CD + c8], h2[(size_t)r * CD + c8] * nrm[e]);
}

// ---------------- final bias + masked gather (preds then targets) ----------------
__global__ void gather_out(const float* __restrict__ agg2, const float* __restrict__ b2,
                           const int* __restrict__ mask, const int* __restrict__ y,
                           float* __restrict__ out) {
    int i = blockIdx.x * 256 + threadIdx.x;
    if (i < NTEST * CD) {
        int t = i >> 3, c = i & 7;
        out[i] = agg2[(size_t)mask[t] * CD + c] + b2[c];
    } else if (i < NTEST * CD + NTEST) {
        int t = i - NTEST * CD;
        out[i] = (float)y[mask[t]];
    }
}

// ---------------- launcher ----------------
extern "C" void kernel_launch(void* const* d_in, const int* in_sizes, int n_in,
                              void* d_out, int out_size, void* d_ws, size_t ws_size,
                              hipStream_t stream) {
    const float* x  = (const float*)d_in[0];
    const int*   ei = (const int*)d_in[1];     // [2, E]: rows then cols
    const float* ew = (const float*)d_in[2];
    const int*   mk = (const int*)d_in[3];
    const int*   y  = (const int*)d_in[4];
    const float* W1 = (const float*)d_in[5];
    const float* b1 = (const float*)d_in[6];
    const float* W2 = (const float*)d_in[7];
    const float* b2 = (const float*)d_in[8];
    float* out = (float*)d_out;

    const int E    = in_sizes[2];              // 320000
    const int Etot = E + NN;                   // with self loops

    // workspace layout (float offsets, 16B aligned)
    float* ws   = (float*)d_ws;
    float* deg  = ws;                          // NN (reused as dis)     [10016]
    float* nrm  = deg + 10016;                 // Etot                   [330000]
    float* h1   = nrm + 330000;                // NN*H1                  [2,000,000]
    float* agg1 = h1 + 2000000;                // NN*H1                  [2,000,000]
    float* h2   = agg1 + 2000000;              // NN*C                   [80,000]
    float* agg2 = h2 + 80000;                  // NN*C                   [80,000]
    unsigned short* Bt = (unsigned short*)(agg2 + 80000);  // NP*KP bf16 (~4.2 MB)

    const int T = 256;
    // 1) normalization
    zero_f32   <<<(NN + T - 1) / T, T, 0, stream>>>(deg, NN);
    deg_kernel <<<(Etot + T - 1) / T, T, 0, stream>>>(ei, ew, deg, E);
    dis_kernel <<<(NN + T - 1) / T, T, 0, stream>>>(deg);
    norm_kernel<<<(Etot + T - 1) / T, T, 0, stream>>>(ei, ew, deg, nrm, E);

    // 2) W1 -> column-major padded bf16
    fill_bt    <<<(NP * KP + T - 1) / T, T, 0, stream>>>(W1, Bt);

    // 3) h1 = x @ W1   (WMMA bf16, f32 accumulate, double-buffered LDS)
    gemm1_wmma <<<NN / 16, 416, 0, stream>>>(x, Bt, h1);

    // 4) layer-1 aggregation
    zero_f32   <<<(NN * H1D + T - 1) / T, T, 0, stream>>>(agg1, NN * H1D);
    scatter200 <<<Etot, T, 0, stream>>>(ei, nrm, h1, agg1, E);

    // 5) h2 = relu(agg1 + b1) @ W2
    gemm2_fused<<<(NN * CD + T - 1) / T, T, 0, stream>>>(agg1, b1, W2, h2);

    // 6) layer-2 aggregation
    zero_f32   <<<(NN * CD + T - 1) / T, T, 0, stream>>>(agg2, NN * CD);
    scatter8   <<<(Etot * CD + T - 1) / T, T, 0, stream>>>(ei, nrm, h2, agg2, E);

    // 7) bias + masked gather (predictions, then targets)
    gather_out <<<(NTEST * (CD + 1) + T - 1) / T, T, 0, stream>>>(agg2, b2, mk, y, out);
}
